// BartCrossAttention_78056735637650
// MI455X (gfx1250) — matrix operand order repaired and checked by reading, not compiled, not run on hardware
//
#include <hip/hip_runtime.h>
#include <hip/hip_bf16.h>

typedef __bf16 bf16_t;
typedef __attribute__((ext_vector_type(16))) __bf16 v16bf;
typedef __attribute__((ext_vector_type(8)))  __bf16 v8bf;
typedef __attribute__((ext_vector_type(4)))  __bf16 v4bf;
typedef __attribute__((ext_vector_type(8)))  float  v8f;

static constexpr int BATCH = 4, LQ = 1024, LKV = 1024, DMODEL = 1024, NH = 16, HD = 64;

// ---------- WMMA helpers (CDNA5 16x16x32 bf16, wave32) ----------

__device__ __forceinline__ v8f wmma_bf16(v16bf a, v16bf b, v8f c) {
  return __builtin_amdgcn_wmma_f32_16x16x32_bf16(false, a, false, b, (short)0, c, false, false);
}

// A-operand fragment: 16x32 bf16. Lane L holds row L%16.
// elements e <- K = base + (L>=16 ? 16:0) + e, fully contiguous.
__device__ __forceinline__ v16bf load_frag_a(const bf16_t* p) {
  v8bf lo = *(const v8bf*)(p);
  v8bf hi = *(const v8bf*)(p + 8);
  return __builtin_shufflevector(lo, hi, 0,1,2,3,4,5,6,7,8,9,10,11,12,13,14,15);
}

// B-operand fragment: 32x16 bf16. Lane L holds column L%16,
// elements e <- K = base + (L>=16 ? 16:0) + e, fully contiguous.
__device__ __forceinline__ v16bf load_frag_b(const bf16_t* p) {
  v8bf lo = *(const v8bf*)(p);
  v8bf hi = *(const v8bf*)(p + 8);
  return __builtin_shufflevector(lo, hi, 0,1,2,3,4,5,6,7,8,9,10,11,12,13,14,15);
}

// ---------- CDNA5 async global->LDS (ASYNCcnt path, ISA 15.18.3 op 98) ----------

__device__ __forceinline__ void async_load_b128(unsigned lds_off, const bf16_t* gptr) {
  asm volatile("global_load_async_to_lds_b128 %0, %1, off"
               :: "v"(lds_off), "v"((unsigned long long)(uintptr_t)gptr)
               : "memory");
}
__device__ __forceinline__ void wait_asynccnt0() {
  asm volatile("s_wait_asynccnt 0" ::: "memory");
}

// ---------- fp32 -> bf16 conversion (4 elems/thread) ----------

__global__ __launch_bounds__(256) void f32_to_bf16_kernel(const float* __restrict__ s,
                                                          bf16_t* __restrict__ d, int n) {
  int i = (blockIdx.x * 256 + threadIdx.x) * 4;
  if (i + 3 < n) {
    float4 f = *(const float4*)(s + i);
    v4bf o;
    o[0] = (bf16_t)f.x; o[1] = (bf16_t)f.y; o[2] = (bf16_t)f.z; o[3] = (bf16_t)f.w;
    *(v4bf*)(d + i) = o;
  }
}

// ---------- GEMM: C[M,N] = A[M,K] * W[N,K]^T + bias ----------
// block = 256 threads = 8 waves (2 M-groups x 4 N-groups), wave tile 32x64,
// block tile 64x256. grid = (N/256, M/64). A-tile (64x32) is double-buffered
// in LDS via async global->LDS loads, shared across the 4 N-wave-groups.

enum { MODE_Q = 0, MODE_KV = 1, MODE_OUT = 2 };

template <int MODE>
__global__ __launch_bounds__(256)
void gemm_wmma_kernel(const bf16_t* __restrict__ A, const bf16_t* __restrict__ W,
                      const float* __restrict__ bias, int K, int N,
                      bf16_t* __restrict__ out_b0, bf16_t* __restrict__ out_b1,
                      float* __restrict__ out_f) {
  __shared__ __align__(16) bf16_t As[2][64 * 32];  // 2 x 4KB double buffer

  const int tid  = threadIdx.x;
  const int lane = tid & 31;
  const int w    = tid >> 5;
  const int col  = lane & 15;
  const bool hiL = lane >= 16;

  const size_t m0 = (size_t)blockIdx.y * 64 + (size_t)(w >> 2) * 32;
  const size_t n0 = (size_t)blockIdx.x * 256 + (size_t)(w & 3) * 64;

  // async-staging assignment: thread -> one 16B chunk of the 64x32 A tile
  const int ar = tid >> 2;            // row in tile 0..63
  const int ac = (tid & 3) * 8;       // k-chunk 0,8,16,24
  const bf16_t* agl = A + ((size_t)blockIdx.y * 64 + ar) * K + ac;
  unsigned lds_dst[2];
  lds_dst[0] = (unsigned)(uintptr_t)(&As[0][ar * 32 + ac]);
  lds_dst[1] = (unsigned)(uintptr_t)(&As[1][ar * 32 + ac]);

  const v8f zf = {0.f,0.f,0.f,0.f,0.f,0.f,0.f,0.f};
  v8f acc[2][4] = {{zf,zf,zf,zf},{zf,zf,zf,zf}};

  const bf16_t* brow[4];
  #pragma unroll
  for (int ni = 0; ni < 4; ++ni)
    brow[ni] = W + (n0 + (size_t)ni * 16 + (size_t)col) * K + (hiL ? 16 : 0);

  const int awrow = (w >> 2) * 32;     // this wave's row group inside the A tile
  const int aoff  = hiL ? 16 : 0;

  async_load_b128(lds_dst[0], agl);    // prologue: stage k-tile 0

  #pragma unroll 2
  for (int k0 = 0; k0 < K; k0 += 32) {
    const int cur = (k0 >> 5) & 1;
    wait_asynccnt0();                  // my wave's async portion landed
    __syncthreads();                   // everyone's landed AND everyone done
                                       // reading the other buffer
    if (k0 + 32 < K) {
      async_load_b128(lds_dst[cur ^ 1], agl + k0 + 32);  // overlap next tile
      __builtin_prefetch(brow[0] + k0 + 64, 0, 1);       // global_prefetch hints
      __builtin_prefetch(brow[2] + k0 + 64, 0, 1);
    }
    v16bf a0 = load_frag_a(&As[cur][(awrow + col) * 32 + aoff]);
    v16bf a1 = load_frag_a(&As[cur][(awrow + 16 + col) * 32 + aoff]);
    #pragma unroll
    for (int ni = 0; ni < 4; ++ni) {
      v16bf b = load_frag_b(brow[ni] + k0);
      acc[0][ni] = wmma_bf16(a0, b, acc[0][ni]);
      acc[1][ni] = wmma_bf16(a1, b, acc[1][ni]);
    }
  }

  const int rowoff = hiL ? 8 : 0;
  #pragma unroll
  for (int mi = 0; mi < 2; ++mi) {
    #pragma unroll
    for (int ni = 0; ni < 4; ++ni) {
      #pragma unroll
      for (int r = 0; r < 8; ++r) {
        const int m = (int)m0 + mi * 16 + r + rowoff;
        const int n = (int)n0 + ni * 16 + col;
        const float v = acc[mi][ni][r] + bias[n];
        if (MODE == MODE_Q) {
          out_b0[(size_t)m * N + n] = (bf16_t)v;
        } else if (MODE == MODE_OUT) {
          out_f[(size_t)m * N + n] = v;
        } else {  // MODE_KV: n = h*128 + s*64 + hd
          const int h = n >> 7, s = (n >> 6) & 1, hd = n & 63;
          const int bb = m >> 10, l = m & 1023;  // LKV = 1024
          if (s == 0) {  // K: [b,h,lk,hd] (hd contiguous)
            out_b0[((size_t)(bb * NH + h) * LKV + l) * HD + hd] = (bf16_t)v;
          } else {       // V transposed: [b,h,hd,lk] (lk contiguous)
            out_b1[((size_t)(bb * NH + h) * HD + hd) * LKV + l] = (bf16_t)v;
          }
        }
      }
    }
  }
}

// ---------- Flash attention: block = 4 waves, each wave = 16 q-rows ----------
// grid = (LQ/64, NH, BATCH). K in [b,h,lk,hd]; V transposed in [b,h,hd,lk]
// so both score and context B-fragments are contiguous global loads.
// Softmax is done in base-2: scores pre-scaled by (1/sqrt(HD))*log2(e), all
// exponentials are bare v_exp_f32 (exp2), running max kept in that domain.

__global__ __launch_bounds__(128)
void attn_wmma_kernel(const bf16_t* __restrict__ Qb, const bf16_t* __restrict__ Kb,
                      const bf16_t* __restrict__ Vtb, bf16_t* __restrict__ Ctx) {
  __shared__ __align__(16) bf16_t Ps[4][16 * 64];   // per-wave P (probs) scratch

  const int lane = threadIdx.x & 31;
  const int w    = threadIdx.x >> 5;
  const int col  = lane & 15;
  const bool hiL = lane >= 16;
  const int rowoff = hiL ? 8 : 0;

  const int b = blockIdx.z, h = blockIdx.y;
  const int q0 = blockIdx.x * 64 + w * 16;

  const bf16_t* Qbase  = Qb  + (size_t)(b * LQ + q0) * DMODEL + h * HD;
  const bf16_t* Kbase  = Kb  + (size_t)(b * NH + h) * LKV * HD;
  const bf16_t* Vtbase = Vtb + (size_t)(b * NH + h) * HD * LKV;

  // Q fragments for this wave's 16 rows (HD=64 -> two 16x32 A frags)
  v16bf aq0 = load_frag_a(Qbase + (size_t)col * DMODEL + 0  + (hiL ? 16 : 0));
  v16bf aq1 = load_frag_a(Qbase + (size_t)col * DMODEL + 32 + (hiL ? 16 : 0));

  const v8f zf = {0.f,0.f,0.f,0.f,0.f,0.f,0.f,0.f};
  v8f o[4] = {zf, zf, zf, zf};   // 16x64 context accumulator (C-layout tiles)
  float m_run[8], l_run[8];
  #pragma unroll
  for (int r = 0; r < 8; ++r) { m_run[r] = -__builtin_inff(); l_run[r] = 0.f; }

  const float scale2 = 0.125f * 1.4426950408889634f;  // (1/sqrt(64)) * log2(e)
  bf16_t* Pw = &Ps[w][0];

  #pragma unroll 2
  for (int k0 = 0; k0 < LKV; k0 += 64) {
    // Scores S = Q (16x64) * K_tile^T -> 4 tiles of 16x16
    v8f s4[4] = {zf, zf, zf, zf};
    #pragma unroll
    for (int kt = 0; kt < 4; ++kt) {
      const bf16_t* kp = Kbase + (size_t)(k0 + kt * 16 + col) * HD + (hiL ? 16 : 0);
      s4[kt] = wmma_bf16(aq0, load_frag_b(kp),      s4[kt]);
      s4[kt] = wmma_bf16(aq1, load_frag_b(kp + 32), s4[kt]);
    }

    // Online softmax (base-2); rows 0-7 in lanes 0-15, rows 8-15 in lanes
    // 16-31: xor-shuffles with mask<16 reduce within each row's lane group.
    #pragma unroll
    for (int r = 0; r < 8; ++r) {
      float mnew = m_run[r];
      #pragma unroll
      for (int kt = 0; kt < 4; ++kt) mnew = fmaxf(mnew, s4[kt][r] * scale2);
      #pragma unroll
      for (int off = 8; off >= 1; off >>= 1) mnew = fmaxf(mnew, __shfl_xor(mnew, off, 32));
      const float corr = exp2f(m_run[r] - mnew);
      m_run[r] = mnew;
      float rs = 0.f;
      #pragma unroll
      for (int kt = 0; kt < 4; ++kt) {
        const float p = exp2f(__builtin_fmaf(s4[kt][r], scale2, -mnew));
        rs += p;
        Pw[(r + rowoff) * 64 + kt * 16 + col] = (bf16_t)p;  // C-layout -> row-major
      }
      #pragma unroll
      for (int off = 8; off >= 1; off >>= 1) rs += __shfl_xor(rs, off, 32);
      l_run[r] = l_run[r] * corr + rs;
      #pragma unroll
      for (int t4 = 0; t4 < 4; ++t4) o[t4][r] *= corr;
    }
    __syncthreads();   // P writes visible for A-layout re-read

    // O += P (16x64) * V_tile (64x64); P from LDS in A-layout,
    // V from transposed global buffer (contiguous per lane).
    #pragma unroll
    for (int kc = 0; kc < 2; ++kc) {
      v16bf ap = load_frag_a(Pw + (size_t)col * 64 + kc * 32 + (hiL ? 16 : 0));
      #pragma unroll
      for (int t4 = 0; t4 < 4; ++t4) {
        const bf16_t* vp = Vtbase + (size_t)(t4 * 16 + col) * LKV
                         + k0 + kc * 32 + (hiL ? 16 : 0);
        o[t4] = wmma_bf16(ap, load_frag_b(vp), o[t4]);
      }
    }
  }

  // Normalize and store context as bf16 [b, lq, h*64+hd]
  #pragma unroll
  for (int r = 0; r < 8; ++r) {
    const float inv = 1.f / l_run[r];
    const size_t rowbase = (size_t)(b * LQ + q0 + r + rowoff) * DMODEL + h * HD;
    #pragma unroll
    for (int t4 = 0; t4 < 4; ++t4)
      Ctx[rowbase + t4 * 16 + col] = (bf16_t)(o[t4][r] * inv);
  }
}

// ---------- launch ----------

extern "C" void kernel_launch(void* const* d_in, const int* in_sizes, int n_in,
                              void* d_out, int out_size, void* d_ws, size_t ws_size,
                              hipStream_t stream) {
  const float* hs  = (const float*)d_in[0];
  const float* kvs = (const float*)d_in[1];
  const float* qw  = (const float*)d_in[2];
  const float* qbi = (const float*)d_in[3];
  const float* kvw = (const float*)d_in[4];
  const float* kvbi= (const float*)d_in[5];
  const float* ow  = (const float*)d_in[6];
  const float* obi = (const float*)d_in[7];
  float* out = (float*)d_out;

  char* ws = (char*)d_ws;
  const size_t MB = 1024 * 1024;
  bf16_t* hsb  = (bf16_t*)(ws + 0);        //  8 MB : hidden_states bf16
  bf16_t* kvsb = (bf16_t*)(ws + 8  * MB);  //  8 MB : key_value_states bf16
  bf16_t* qwb  = (bf16_t*)(ws + 16 * MB);  //  2 MB : q_weight bf16
  bf16_t* kvwb = (bf16_t*)(ws + 18 * MB);  //  4 MB : kv_weight bf16
  bf16_t* owb  = (bf16_t*)(ws + 22 * MB);  //  2 MB : out_weight bf16
  bf16_t* Qb   = (bf16_t*)(ws + 24 * MB);  //  8 MB : Q   [b,lq,1024]
  bf16_t* Kb   = (bf16_t*)(ws + 32 * MB);  //  8 MB : K   [b,h,lk,64]
  bf16_t* Vtb  = (bf16_t*)(ws + 40 * MB);  //  8 MB : V^T [b,h,64,lk]
  bf16_t* Ctx  = (bf16_t*)(ws + 48 * MB);  //  8 MB : context [b,lq,1024]

  auto cvt = [&](const float* s, bf16_t* d, int n) {
    f32_to_bf16_kernel<<<dim3(n / 1024), 256, 0, stream>>>(s, d, n);
  };
  cvt(hs,  hsb,  BATCH * LQ  * DMODEL);
  cvt(kvs, kvsb, BATCH * LKV * DMODEL);
  cvt(qw,  qwb,  DMODEL * DMODEL);
  cvt(kvw, kvwb, 2 * DMODEL * DMODEL);
  cvt(ow,  owb,  DMODEL * DMODEL);

  // Q = hs @ q_w^T + q_b  (M=4096, N=1024, K=1024)
  gemm_wmma_kernel<MODE_Q><<<dim3(1024 / 256, 4096 / 64), 256, 0, stream>>>(
      hsb, qwb, qbi, DMODEL, DMODEL, Qb, nullptr, nullptr);
  // KV = kvs @ kv_w^T + kv_b, K head-major + V transposed (N=2048)
  gemm_wmma_kernel<MODE_KV><<<dim3(2048 / 256, 4096 / 64), 256, 0, stream>>>(
      kvsb, kvwb, kvbi, DMODEL, 2 * DMODEL, Kb, Vtb, nullptr);
  // Flash attention -> Ctx
  attn_wmma_kernel<<<dim3(LQ / 64, NH, BATCH), 128, 0, stream>>>(Qb, Kb, Vtb, Ctx);
  // out = Ctx @ out_w^T + out_b -> f32 d_out
  gemm_wmma_kernel<MODE_OUT><<<dim3(1024 / 256, 4096 / 64), 256, 0, stream>>>(
      Ctx, owb, obi, DMODEL, DMODEL, nullptr, nullptr, out);
}
